// RUMLayer_77008763617442
// MI455X (gfx1250) — matrix operand order repaired and checked by reading, not compiled, not run on hardware
//
#include <hip/hip_runtime.h>
#include <cstdint>
#include <cstddef>

typedef __attribute__((ext_vector_type(16))) __bf16 v16bf;
typedef __attribute__((ext_vector_type(8)))  float  v8f;
typedef __attribute__((ext_vector_type(8)))  unsigned short u16x8;

#define H      128
#define L      12
#define G3     384            // 3*H
#define TILE   16             // walks per workgroup (== WMMA M)
#define NWALK  65536          // S*E
#define NTILES (NWALK / TILE)
#define NSS    100

// ---- LDS layout (bytes), all offsets 128-aligned ----
#define O_RW   0              // int[16*12]           768
#define O_DEG  768            // float[16*12]         768
#define O_RK   1536           // int[16*12]           768 (anonymised ranks)
#define O_HF   3072           // float[16*128]       8192
#define O_H0   11264          // float[16*128]       8192
#define O_GI   19456          // float[16*384]      24576
#define O_GH   44032          // float[16*384]      24576
#define O_HA   68608          // u16[2048] packed A  4096
#define O_XA   72704          // u16[2048] packed A  4096 (x gather, buffer 0)
#define O_YF   76800          // u16[12*2048]       49152
#define O_YB   125952         // u16[12*2048]       49152
#define O_SEL  175104         // int[17]              128
#define O_XB   175232         // u16[2048] packed A  4096 (x gather, buffer 1)
#define SMEM_BYTES 179328

__device__ inline uint16_t f2bf(float f) {
  union { float f; uint32_t u; } v; v.f = f;
  uint32_t r = v.u + 0x7FFFu + ((v.u >> 16) & 1u);   // round-to-nearest-even
  return (uint16_t)(r >> 16);
}
__device__ inline float fast_rcp(float x) {
#if __has_builtin(__builtin_amdgcn_rcpf)
  return __builtin_amdgcn_rcpf(x);
#else
  return 1.f / x;
#endif
}
__device__ inline float fast_tanh(float x) {
#if __has_builtin(__builtin_amdgcn_tanhf)
  return __builtin_amdgcn_tanhf(x);
#else
  return tanhf(x);
#endif
}
__device__ inline float sigmoidf_(float x) { return fast_rcp(1.f + __expf(-x)); }
__device__ inline float softplusf_(float x) { return fmaxf(x, 0.f) + log1pf(__expf(-fabsf(x))); }

// A-fragment packed layout for a 16x128 bf16 tile (contiguous k order: kk = 16*(lane/16) + slot, matching ldB):
// chunk c = j>>5; kk=j&31; lane = m + 16*(kk>>4); slot = kk&15
// NOTE: an aligned run of 8 consecutive j maps to 8 contiguous slots (16B) -> b128 LDS ops.
__device__ inline int packA_idx(int m, int j) {
  int c = j >> 5, kk = j & 31;
  int lane = m + ((kk >> 4) << 4);
  int slot = kk & 15;
  return ((c << 5) + lane) * 16 + slot;
}
__device__ inline v16bf ldA(const uint16_t* s_a, int c, int lane) {
  return *(const v16bf*)(s_a + (((c << 5) + lane) << 4));
}
// B fragments: row-major W[g][k] bf16; lane holds col n=g, 16 contiguous K
__device__ inline v16bf ldB(const uint16_t* w, int g, int c, int nhalf) {
  return *(const v16bf*)(w + g * H + (nhalf << 4) + (c << 5));
}
// preload this wave's 3 N-tiles x 4 K-chunks of B into registers (96 VGPRs/lane)
__device__ inline void ldB_frags(const uint16_t* W, int wave, int lane, v16bf* bf) {
  const int nhalf = lane >> 4, n = lane & 15;
#pragma unroll
  for (int q = 0; q < 3; ++q)
#pragma unroll
    for (int c = 0; c < 4; ++c)
      bf[q * 4 + c] = ldB(W, (wave * 3 + q) * 16 + n, c, nhalf);
}

// gh[m][g] = (h @ Whh^T)[m][g] + bhh[g]  with register-resident B fragments
__device__ inline void wmma_gh_reg(const uint16_t* s_a, const v16bf* bf,
                                   const float* bias, float* s_out, int wave, int lane) {
  const int nhalf = lane >> 4, n = lane & 15;
#pragma unroll
  for (int q = 0; q < 3; ++q) {
    int g = (wave * 3 + q) * 16 + n;
    v8f acc = {};
#pragma unroll
    for (int c = 0; c < 4; ++c)
      acc = __builtin_amdgcn_wmma_f32_16x16x32_bf16(false, ldA(s_a, c, lane), false, bf[q * 4 + c], (short)0, acc, false, false);
    float bb = bias[g];
#pragma unroll
    for (int v = 0; v < 8; ++v)
      s_out[(v + (nhalf << 3)) * G3 + g] = acc[v] + bb;
  }
}

// gi[m][g] = xf@Wx^T + yf@Wyf^T + yb@Wyb^T + bih[g] + deg[m,t]*wdeg[g]  (B streamed from L2)
__device__ inline void wmma_gi(const uint16_t* s_xa, const uint16_t* s_yfa, const uint16_t* s_yba,
                               const uint16_t* Wx, const uint16_t* Wyf, const uint16_t* Wyb,
                               const float* bias, const float* wdeg, const float* s_deg, int t,
                               float* s_out, int wave, int lane) {
  const int nhalf = lane >> 4, n = lane & 15;
#pragma unroll 1
  for (int q = 0; q < 3; ++q) {
    int g = (wave * 3 + q) * 16 + n;
    v8f acc = {};
#pragma unroll
    for (int c = 0; c < 4; ++c)
      acc = __builtin_amdgcn_wmma_f32_16x16x32_bf16(false, ldA(s_xa, c, lane),  false, ldB(Wx,  g, c, nhalf), (short)0, acc, false, false);
#pragma unroll
    for (int c = 0; c < 4; ++c)
      acc = __builtin_amdgcn_wmma_f32_16x16x32_bf16(false, ldA(s_yfa, c, lane), false, ldB(Wyf, g, c, nhalf), (short)0, acc, false, false);
#pragma unroll
    for (int c = 0; c < 4; ++c)
      acc = __builtin_amdgcn_wmma_f32_16x16x32_bf16(false, ldA(s_yba, c, lane), false, ldB(Wyb, g, c, nhalf), (short)0, acc, false, false);
    float bb = bias[g], wd = wdeg[g];
#pragma unroll
    for (int v = 0; v < 8; ++v) {
      int m = v + (nhalf << 3);
      s_out[m * G3 + g] = acc[v] + bb + s_deg[m * L + t] * wd;
    }
  }
}

// per-lane async DMA gather: one 16B x-row slice, global -> packed-A LDS (ASYNCcnt)
__device__ inline void async_gather_x(const uint16_t* x_bf, const int* s_rw, int t,
                                      uint16_t* s_dst, int tid) {
  int it0 = tid << 3;
  int m = it0 >> 7, j0 = it0 & 127;
  int node = s_rw[m * L + t];
  uint32_t goff = ((uint32_t)node << 8) + (uint32_t)(j0 << 1);          // bytes into x_bf
  uint32_t lds  = (uint32_t)(size_t)(s_dst + packA_idx(m, j0));         // LDS byte address
  asm volatile("global_load_async_to_lds_b128 %0, %1, %2"
               :: "v"(lds), "v"(goff), "s"(x_bf) : "memory");
}
__device__ inline void wait_async0() {
  asm volatile("s_wait_asynccnt 0" ::: "memory");
}

__device__ inline void ld8(const float* p, float* d) {   // two float4 loads
  float4 a = *(const float4*)p;
  float4 b = *(const float4*)(p + 4);
  d[0] = a.x; d[1] = a.y; d[2] = a.z; d[3] = a.w;
  d[4] = b.x; d[5] = b.y; d[6] = b.z; d[7] = b.w;
}

// GRU gate math over an aligned run of 8 hidden units
__device__ inline void gates_core(const float* ir, const float* iz, const float* in,
                                  const float* hr, const float* hz, const float* hnv,
                                  float* s_hf, int it0, float* hn, u16x8* hb) {
#pragma unroll
  for (int k = 0; k < 8; ++k) {
    float r  = sigmoidf_(ir[k] + hr[k]);
    float z  = sigmoidf_(iz[k] + hz[k]);
    float nn = fast_tanh(in[k] + r * hnv[k]);
    float h  = s_hf[it0 + k];
    hn[k] = (1.f - z) * nn + z * h;
    s_hf[it0 + k] = hn[k];
    (*hb)[k] = f2bf(hn[k]);
  }
}

// small-GRU gate update: gi read from 12x384 rank table (global, L2-resident)
__device__ inline void gates_update_tab(const float* giT, const int* s_rk, int t,
                                        const float* s_gh, float* s_hf,
                                        uint16_t* s_ha, uint16_t* s_yp, int tid) {
  const int it0 = tid << 3;
  const int m = it0 >> 7, j0 = it0 & 127;
  const float* gi = giT + (size_t)s_rk[m * L + t] * G3 + j0;
  const int b = m * G3 + j0;
  float ir[8], iz[8], in[8], hr[8], hz[8], hnv[8], hn[8];
  ld8(gi, ir); ld8(gi + H, iz); ld8(gi + 2 * H, in);
  ld8(s_gh + b, hr); ld8(s_gh + b + H, hz); ld8(s_gh + b + 2 * H, hnv);
  u16x8 hb;
  gates_core(ir, iz, in, hr, hz, hnv, s_hf, it0, hn, &hb);
  const int pk = packA_idx(m, j0);
  *(u16x8*)(s_ha + pk) = hb;
  *(u16x8*)(s_yp + pk) = hb;
}

// main-GRU gate update: gi from LDS (WMMA output); also exports selected y rows
__device__ inline void gates_update_main(const float* s_gi, const float* s_gh, float* s_hf,
                                         uint16_t* s_ha, float* y_sel, int t,
                                         const int* s_sel, int tid) {
  const int it0 = tid << 3;
  const int m = it0 >> 7, j0 = it0 & 127;
  const int b = m * G3 + j0;
  float ir[8], iz[8], in[8], hr[8], hz[8], hnv[8], hn[8];
  ld8(s_gi + b, ir); ld8(s_gi + b + H, iz); ld8(s_gi + b + 2 * H, in);
  ld8(s_gh + b, hr); ld8(s_gh + b + H, hz); ld8(s_gh + b + 2 * H, hnv);
  u16x8 hb;
  gates_core(ir, iz, in, hr, hz, hnv, s_hf, it0, hn, &hb);
  *(u16x8*)(s_ha + packA_idx(m, j0)) = hb;
  if (t < L - 1) {
    int ns = s_sel[0]; if (ns > 8) ns = 8;
    for (int q = 0; q < ns; ++q)
      if (s_sel[9 + q] == m) {
        float* dst = y_sel + (s_sel[1 + q] * (L - 1) + t) * H + j0;
#pragma unroll
        for (int k = 0; k < 8; ++k) dst[k] = hn[k];
      }
  }
}

// ---------------- prep: bf16 repack, x->bf16, gi rank tables, degree max ----------------
__global__ __launch_bounds__(256) void rum_prep(
    const float* __restrict__ whhf, const float* __restrict__ whhb, const float* __restrict__ whhm,
    const float* __restrict__ wihm, const float* __restrict__ x,
    const float* __restrict__ wihf2, const float* __restrict__ bihf,
    const float* __restrict__ wihb2, const float* __restrict__ bihb,
    const int* __restrict__ walks, const int* __restrict__ dstdeg,
    uint16_t* WhhF, uint16_t* WhhB, uint16_t* WhhM,
    uint16_t* Wx, uint16_t* Wyf, uint16_t* Wyb, float* wdeg,
    float* giTf, float* giTb,
    uint16_t* x_bf, int nx, int* degmax) {
  const int i0 = blockIdx.x * 256 + threadIdx.x;
  const int nthr = gridDim.x * 256;
  if (i0 < G3 * H) {
    WhhF[i0] = f2bf(whhf[i0]); WhhB[i0] = f2bf(whhb[i0]); WhhM[i0] = f2bf(whhm[i0]);
    int g = i0 >> 7, k = i0 & 127;
    const float* row = wihm + g * 385;
    Wx[i0]  = f2bf(row[k]);
    Wyf[i0] = f2bf(row[128 + k]);
    Wyb[i0] = f2bf(row[256 + k]);
    if (k == 0) wdeg[g] = row[384];
  }
  if (i0 < L * G3) {                       // 12 rank rows x 384 gates
    int r = i0 / G3, g = i0 - r * G3;
    float ang = (float)r * (6.28318530718f / (float)L);
    float s = sinf(ang), c = cosf(ang);
    giTf[i0] = s * wihf2[g * 2] + c * wihf2[g * 2 + 1] + bihf[g];
    giTb[i0] = s * wihb2[g * 2] + c * wihb2[g * 2 + 1] + bihb[g];
  }
  for (int j = i0; j < nx; j += nthr) x_bf[j] = f2bf(x[j]);
  int lmax = 0;
  for (int j = i0; j < NWALK * L; j += nthr) {
    int d = dstdeg[walks[j]];
    if (d > lmax) lmax = d;
  }
  __shared__ int sred[256];
  sred[threadIdx.x] = lmax;
  __syncthreads();
  for (int s2 = 128; s2 > 0; s2 >>= 1) {
    if (threadIdx.x < s2 && sred[threadIdx.x + s2] > sred[threadIdx.x]) sred[threadIdx.x] = sred[threadIdx.x + s2];
    __syncthreads();
  }
  if (threadIdx.x == 0) atomicMax(degmax, sred[0]);
}

// ---------------- main fused 3-GRU kernel: one 16-walk tile per workgroup ----------------
__global__ __launch_bounds__(256) void rum_main(
    const uint16_t* __restrict__ x_bf, const int* __restrict__ walks,
    const int* __restrict__ dstdeg, const int* __restrict__ ss,
    const float* __restrict__ giTf, const float* __restrict__ bhhf,
    const float* __restrict__ giTb, const float* __restrict__ bhhb,
    const float* __restrict__ bihm, const float* __restrict__ bhhm, const float* __restrict__ wdeg,
    const uint16_t* __restrict__ WhhF, const uint16_t* __restrict__ WhhB, const uint16_t* __restrict__ WhhM,
    const uint16_t* __restrict__ Wx, const uint16_t* __restrict__ Wyf, const uint16_t* __restrict__ Wyb,
    const int* __restrict__ degmax, float* __restrict__ y_sel, float* __restrict__ out_h) {
  extern __shared__ __align__(128) char smem[];
  int*      s_rw  = (int*)(smem + O_RW);
  float*    s_deg = (float*)(smem + O_DEG);
  int*      s_rk  = (int*)(smem + O_RK);
  float*    s_hf  = (float*)(smem + O_HF);
  float*    s_h0  = (float*)(smem + O_H0);
  float*    s_gi  = (float*)(smem + O_GI);
  float*    s_gh  = (float*)(smem + O_GH);
  uint16_t* s_ha  = (uint16_t*)(smem + O_HA);
  uint16_t* s_xa  = (uint16_t*)(smem + O_XA);
  uint16_t* s_xb  = (uint16_t*)(smem + O_XB);
  uint16_t* s_yf  = (uint16_t*)(smem + O_YF);
  uint16_t* s_yb  = (uint16_t*)(smem + O_YB);
  int*      s_sel = (int*)(smem + O_SEL);

  const int tid  = threadIdx.x;
  const int lane = tid & 31, wave = tid >> 5;
  const int tile = blockIdx.x;

  if (tid == 0) s_sel[0] = 0;
  // per-walk: anonymise ranks, reversed node ids, normalized degrees
  if (tid < TILE) {
    int m = tid, r = tile * TILE + m;
    int a[L];
#pragma unroll
    for (int i = 0; i < L; ++i) a[i] = walks[r * L + i];
    int dm = degmax[0]; if (dm < 1) dm = 1;
    float inv_dm = 1.f / (float)dm;
    int firstrank[L]; int rank = 0;
    for (int i = 0; i < L; ++i) {
      int f = 0; while (a[f] != a[i]) ++f;
      if (f == i) firstrank[i] = rank++;
      int t = L - 1 - i;                           // reversed position
      s_rk[m * L + t]  = firstrank[f];
      s_rw[m * L + t]  = a[i];
      s_deg[m * L + t] = (float)dstdeg[a[i]] * inv_dm;
      __builtin_prefetch(x_bf + (size_t)a[i] * H, 0, 1);  // warm x rows for main phase
    }
  }
  { // zero h (f32 + packed bf16), vectorized runs of 8
    int it0 = tid << 3;
    float4 z4 = make_float4(0.f, 0.f, 0.f, 0.f);
    *(float4*)(s_hf + it0) = z4; *(float4*)(s_hf + it0 + 4) = z4;
    u16x8 z8 = (u16x8)0;
    *(u16x8*)(s_ha + it0) = z8;
  }
  __syncthreads();
  // which of the 100 supervised walks live in this tile
  for (int p = tid; p < NSS; p += 256) {
    int r = ss[p];
    if ((r >> 4) == tile) {
      int slot = atomicAdd(&s_sel[0], 1);
      if (slot < 8) { s_sel[1 + slot] = p; s_sel[9 + slot] = r & 15; }
    }
  }
  __syncthreads();

  // ---- forward tiny GRU (Whh B fragments live in registers; gi from rank table) ----
  {
    v16bf bW[12];
    ldB_frags(WhhF, wave, lane, bW);
#pragma unroll 1
    for (int t = 0; t < L; ++t) {
      wmma_gh_reg(s_ha, bW, bhhf, s_gh, wave, lane);
      __syncthreads();
      gates_update_tab(giTf, s_rk, t, s_gh, s_hf, s_ha, s_yf + t * 2048, tid);
      __syncthreads();
    }
  }
  { // save 0.5*hf, reset h
    int it0 = tid << 3;
#pragma unroll
    for (int k = 0; k < 8; ++k) { s_h0[it0 + k] = 0.5f * s_hf[it0 + k]; s_hf[it0 + k] = 0.f; }
    u16x8 z8 = (u16x8)0;
    *(u16x8*)(s_ha + it0) = z8;
  }
  __syncthreads();

  // ---- backward tiny GRU (input time-reversed: rank index L-1-t) ----
  {
    v16bf bW[12];
    ldB_frags(WhhB, wave, lane, bW);
#pragma unroll 1
    for (int t = 0; t < L; ++t) {
      wmma_gh_reg(s_ha, bW, bhhb, s_gh, wave, lane);
      __syncthreads();
      gates_update_tab(giTb, s_rk, L - 1 - t, s_gh, s_hf, s_ha, s_yb + t * 2048, tid);
      __syncthreads();
    }
  }
  { // h0 = 0.5*(hf+hb) into live h (f32 + packed bf16)
    int it0 = tid << 3;
    int m = it0 >> 7, j0 = it0 & 127;
    u16x8 hb;
#pragma unroll
    for (int k = 0; k < 8; ++k) {
      float hv = s_h0[it0 + k] + 0.5f * s_hf[it0 + k];
      s_hf[it0 + k] = hv;
      hb[k] = f2bf(hv);
    }
    *(u16x8*)(s_ha + packA_idx(m, j0)) = hb;
  }
  __syncthreads();

  // ---- main GRU: feat = [x[w], yf_t, yb_rev_t, deg]; x gathered by async DMA, 2-deep pipeline ----
  {
    v16bf bW[12];
    ldB_frags(WhhM, wave, lane, bW);
    uint16_t* s_x2[2] = { s_xa, s_xb };
    async_gather_x(x_bf, s_rw, 0, s_x2[0], tid);       // prologue: gather t=0
#pragma unroll 1
    for (int t = 0; t < L; ++t) {
      wmma_gh_reg(s_ha, bW, bhhm, s_gh, wave, lane);   // overlaps in-flight gather
      wait_async0();
      __syncthreads();                                  // x buffer t visible to all waves
      if (t + 1 < L) async_gather_x(x_bf, s_rw, t + 1, s_x2[(t + 1) & 1], tid);
      wmma_gi(s_x2[t & 1], s_yf + t * 2048, s_yb + (L - 1 - t) * 2048,
              Wx, Wyf, Wyb, bihm, wdeg, s_deg, t, s_gi, wave, lane);
      __syncthreads();
      gates_update_main(s_gi, s_gh, s_hf, s_ha, y_sel, t, s_sel, tid);
      __syncthreads();
    }
  }
  // hT out (contiguous, vectorizable)
  {
    int it0 = tid << 3;
    float* dst = out_h + (size_t)tile * (TILE * H) + it0;
    *(float4*)dst       = *(const float4*)(s_hf + it0);
    *(float4*)(dst + 4) = *(const float4*)(s_hf + it0 + 4);
  }
}

// ---------------- loss epilogue: 100 selected walks ----------------
__global__ __launch_bounds__(256) void rum_loss(
    const float* __restrict__ y0, const float* __restrict__ fcw, const float* __restrict__ fcb,
    const int* __restrict__ walks, const int* __restrict__ ss,
    const float* __restrict__ y_sel, float* __restrict__ out_loss) {
  __shared__ float red[256];
  __shared__ float s_pw;
  const int tid = threadIdx.x;
  const int TOT = NSS * (L - 1) * 64;
  float sum = 0.f;
  for (int it = tid; it < TOT; it += 256) {
    int f = it & 63, pt = it >> 6, tt = pt % (L - 1), p = pt / (L - 1);
    int node = walks[ss[p] * L + (L - 2) - tt];       // w[r][tt+1] = walks[r][11-(tt+1)]
    sum += y0[node * 64 + f];
  }
  red[tid] = sum; __syncthreads();
  for (int s2 = 128; s2 > 0; s2 >>= 1) { if (tid < s2) red[tid] += red[tid + s2]; __syncthreads(); }
  if (tid == 0) s_pw = (float)TOT / red[0];           // pw = 1/mean(yt)
  __syncthreads();
  float pw = s_pw;
  float ls = 0.f;
  for (int it = tid; it < TOT; it += 256) {
    int f = it & 63, pt = it >> 6, tt = pt % (L - 1), p = pt / (L - 1);
    int node = walks[ss[p] * L + (L - 2) - tt];
    float yt = y0[node * 64 + f];
    const float* yh = y_sel + (p * (L - 1) + tt) * H;
    const float* wr = fcw + f * H;
    float lg = fcb[f];
    for (int j = 0; j < H; ++j) lg += yh[j] * wr[j];
    ls += pw * yt * softplusf_(-lg) + (1.f - yt) * softplusf_(lg);
  }
  red[tid] = ls; __syncthreads();
  for (int s2 = 128; s2 > 0; s2 >>= 1) { if (tid < s2) red[tid] += red[tid + s2]; __syncthreads(); }
  if (tid == 0) out_loss[0] = red[0] / (float)TOT;
}

extern "C" void kernel_launch(void* const* d_in, const int* in_sizes, int n_in,
                              void* d_out, int out_size, void* d_ws, size_t ws_size,
                              hipStream_t stream) {
  (void)n_in; (void)out_size; (void)ws_size;
  const float* x    = (const float*)d_in[0];
  const float* y0   = (const float*)d_in[1];
  const float* wihf = (const float*)d_in[2];
  const float* whhf = (const float*)d_in[3];
  const float* bihf = (const float*)d_in[4];
  const float* bhhf = (const float*)d_in[5];
  const float* wihb = (const float*)d_in[6];
  const float* whhb = (const float*)d_in[7];
  const float* bihb = (const float*)d_in[8];
  const float* bhhb = (const float*)d_in[9];
  const float* wihm = (const float*)d_in[10];
  const float* whhm = (const float*)d_in[11];
  const float* bihm = (const float*)d_in[12];
  const float* bhhm = (const float*)d_in[13];
  const float* fcw  = (const float*)d_in[14];
  const float* fcb  = (const float*)d_in[15];
  const int*   walks  = (const int*)d_in[16];
  const int*   dstdeg = (const int*)d_in[17];
  const int*   ss     = (const int*)d_in[18];
  const int    nx     = in_sizes[0];                 // N * NIN

  char* ws = (char*)d_ws;
  const size_t WB = (size_t)G3 * H * sizeof(uint16_t);   // 98304 B per bf16 weight block
  int*      degmax = (int*)ws;
  uint16_t* WhhF = (uint16_t*)(ws + 256);
  uint16_t* WhhB = (uint16_t*)(ws + 256 + 1 * WB);
  uint16_t* WhhM = (uint16_t*)(ws + 256 + 2 * WB);
  uint16_t* Wx   = (uint16_t*)(ws + 256 + 3 * WB);
  uint16_t* Wyf  = (uint16_t*)(ws + 256 + 4 * WB);
  uint16_t* Wyb  = (uint16_t*)(ws + 256 + 5 * WB);
  size_t off = 256 + 6 * WB;                             // 590080
  float*    wdeg = (float*)(ws + off);  off += 2048;     // 384 f32 (padded)
  float*    giTf = (float*)(ws + off);  off += (size_t)L * G3 * 4;  // 12x384 f32
  float*    giTb = (float*)(ws + off);  off += (size_t)L * G3 * 4;
  float*    y_sel= (float*)(ws + off);  off += (size_t)NSS * (L - 1) * H * 4;
  uint16_t* x_bf = (uint16_t*)(ws + off);                // N*128 bf16
  float*    out_h    = (float*)d_out;
  float*    out_loss = out_h + (size_t)NWALK * H;

  hipMemsetAsync(degmax, 0, sizeof(int), stream);
  rum_prep<<<1024, 256, 0, stream>>>(whhf, whhb, whhm, wihm, x,
                                     wihf, bihf, wihb, bihb, walks, dstdeg,
                                     WhhF, WhhB, WhhM, Wx, Wyf, Wyb, wdeg,
                                     giTf, giTb, x_bf, nx, degmax);
  hipFuncSetAttribute((const void*)rum_main, hipFuncAttributeMaxDynamicSharedMemorySize, SMEM_BYTES);
  rum_main<<<NTILES, 256, SMEM_BYTES, stream>>>(
      x_bf, walks, dstdeg, ss,
      giTf, bhhf, giTb, bhhb, bihm, bhhm, wdeg,
      WhhF, WhhB, WhhM, Wx, Wyf, Wyb, degmax, y_sel, out_h);
  rum_loss<<<1, 256, 0, stream>>>(y0, fcw, fcb, walks, ss, y_sel, out_loss);
}
